// Seq2Seq_21131239096579
// MI455X (gfx1250) — matrix operand order repaired and checked by reading, not compiled, not run on hardware
//
#include <hip/hip_runtime.h>

typedef __attribute__((ext_vector_type(16))) __bf16 v16bf;
typedef __attribute__((ext_vector_type(8)))  __bf16 v8bf;
typedef __attribute__((ext_vector_type(8)))  float  v8f;

#define BATCH   32
#define SRC_LEN 64
#define TGT_LEN 64
#define VOCAB   32000
#define EDIM    512
#define HDIM    1024
#define KDIM    1536   /* E + H */
#define G4H     4096   /* 4*H   */
#define SOS     1

__device__ __forceinline__ float sigmoidf(float x) { return 1.0f / (1.0f + __expf(-x)); }

// ---- WMMA fragment loaders (wave32 layouts per CDNA5 ISA 7.12.2) ----
// A 16x32 bf16: lane holds row M=lane%16; elem i -> K = (i<8 ? half*8+i : 16+half*8+(i-8))
__device__ __forceinline__ v16bf load_a_f32(const float* __restrict__ rp, int half) {
  const float* p0 = rp + half * 8;
  const float* p1 = rp + 16 + half * 8;
  v16bf a;
#pragma unroll
  for (int i = 0; i < 8; ++i) { a[i] = (__bf16)p0[i]; a[i + 8] = (__bf16)p1[i]; }
  return a;
}
__device__ __forceinline__ v16bf load_a_bf(const __bf16* __restrict__ rp, int half) {
  v8bf lo = *(const v8bf*)(rp + half * 8);
  v8bf hi = *(const v8bf*)(rp + 16 + half * 8);
  v16bf a;
#pragma unroll
  for (int i = 0; i < 8; ++i) { a[i] = lo[i]; a[i + 8] = hi[i]; }
  return a;
}
// B 32x16 bf16: lane holds col N=lane%16; elem i -> K = (i<8 ? half*8+i : 16+half*8+(i-8))
__device__ __forceinline__ v16bf load_b_bf(const __bf16* __restrict__ rp, int half) {
  v8bf lo = *(const v8bf*)(rp + half * 8);
  v8bf hi = *(const v8bf*)(rp + 16 + half * 8);
  v16bf b;
#pragma unroll
  for (int i = 0; i < 8; ++i) { b[i] = lo[i]; b[i + 8] = hi[i]; }
  return b;
}

// 4 gate B-tiles x 2 M-tiles of WMMA for one K-chunk
__device__ __forceinline__ void gates_mma(const __bf16* __restrict__ W, int n0l16, int kbase,
                                          int half, const v16bf (&afrag)[2], v8f (&acc)[2][4]) {
#pragma unroll
  for (int g = 0; g < 4; ++g) {
    v16bf bfrag = load_b_bf(W + (size_t)(g * HDIM + n0l16) * KDIM + kbase, half);
#pragma unroll
    for (int mt = 0; mt < 2; ++mt)
      acc[mt][g] = __builtin_amdgcn_wmma_f32_16x16x32_bf16(
          false, afrag[mt], false, bfrag, (short)0, acc[mt][g], false, false);
  }
}

// ---- LSTM step: gates = [x|h] @ [Wih|Whh]^T + b ; update c, h ----
// grid 64 blocks (16 cols of each gate), 4 waves K-split 1536/4, LDS reduce, fused activation.
// emb/h boundary is wave-uniform: split into two scalar-trip-count loops (no exec divergence).
__global__ __launch_bounds__(128) void lstm_step_kernel(
    const float* __restrict__ emb,          // [V, 512] f32
    const int* __restrict__ tok_idx, int tok_stride,
    const __bf16* __restrict__ W,           // [4096, 1536] bf16 (combined Wih|Whh)
    const float* __restrict__ bias,         // [4096] f32 (bih+bhh)
    const __bf16* __restrict__ h_in,        // [32, 1024] bf16
    __bf16* __restrict__ h_out,             // [32, 1024] bf16
    float* __restrict__ c_st)               // [32, 1024] f32 (in place)
{
  const int lane = threadIdx.x & 31;
  const int wave = __builtin_amdgcn_readfirstlane(threadIdx.x >> 5);  // SGPR-uniform
  const int half = lane >> 4;
  const int l16  = lane & 15;
  const int n0   = blockIdx.x * 16;

  const int kb0 = wave * 384;
  int ne = (kb0 < EDIM) ? ((EDIM - kb0) >> 5) : 0;  // # embedding chunks for this wave
  if (ne > 12) ne = 12;                             // wave0: 12, wave1: 4, waves2-3: 0

  // hoist token gather + embedding row base pointers (one load per M-tile)
  const float* erow[2];
#pragma unroll
  for (int mt = 0; mt < 2; ++mt) {
    const int tok = tok_idx[(mt * 16 + l16) * tok_stride];
    erow[mt] = emb + (size_t)tok * EDIM;
  }

  // warm L2/WGP$ for this wave's weight rows
#pragma unroll
  for (int g = 0; g < 4; ++g)
    __builtin_prefetch(W + (size_t)(g * HDIM + n0 + l16) * KDIM + kb0, 0, 3);

  v8f acc[2][4] = {};
  // embedding-region chunks (f32 gather -> bf16 fragments)
  for (int kc = 0; kc < ne; ++kc) {
    const int kbase = kb0 + kc * 32;
    v16bf afrag[2];
#pragma unroll
    for (int mt = 0; mt < 2; ++mt) afrag[mt] = load_a_f32(erow[mt] + kbase, half);
    gates_mma(W, n0 + l16, kbase, half, afrag, acc);
  }
  // h-region chunks (bf16 fragments)
  for (int kc = ne; kc < 12; ++kc) {
    const int kbase = kb0 + kc * 32;
    v16bf afrag[2];
#pragma unroll
    for (int mt = 0; mt < 2; ++mt)
      afrag[mt] = load_a_bf(h_in + (mt * 16 + l16) * HDIM + (kbase - EDIM), half);
    gates_mma(W, n0 + l16, kbase, half, afrag, acc);
  }

  __shared__ float red[4][2][4][8][32];
#pragma unroll
  for (int mt = 0; mt < 2; ++mt)
#pragma unroll
    for (int g = 0; g < 4; ++g)
#pragma unroll
      for (int v = 0; v < 8; ++v)
        red[wave][mt][g][v][lane] = acc[mt][g][v];
  __syncthreads();
  if (wave != 0) return;

  const int col = n0 + l16;
#pragma unroll
  for (int mt = 0; mt < 2; ++mt) {
#pragma unroll
    for (int v = 0; v < 8; ++v) {
      const int row = mt * 16 + 8 * half + v;    // C/D layout: M = v + 8*(lane/16)
      float gs[4];
#pragma unroll
      for (int g = 0; g < 4; ++g)
        gs[g] = acc[mt][g][v] + red[1][mt][g][v][lane] + red[2][mt][g][v][lane] +
                red[3][mt][g][v][lane] + bias[g * HDIM + col];
      const float c_old = c_st[row * HDIM + col];
      const float c_new = sigmoidf(gs[1]) * c_old + sigmoidf(gs[0]) * tanhf(gs[2]);
      const float h_new = sigmoidf(gs[3]) * tanhf(c_new);
      c_st[row * HDIM + col]  = c_new;
      h_out[row * HDIM + col] = (__bf16)h_new;
    }
  }
}

// ---- decoder fc: logits[b, t, :] = h @ fc_W^T + fc_b ; 500 blocks x 4 waves, 64 vocab rows/block
__global__ __launch_bounds__(128) void fc_logits_kernel(
    const __bf16* __restrict__ h_bf,        // [32, 1024] bf16
    const __bf16* __restrict__ fcW,         // [32000, 1024] bf16
    const float* __restrict__ fc_b,         // [32000]
    float* __restrict__ out, int t)         // out[b, t, v] f32
{
  const int lane = threadIdx.x & 31;
  const int wave = __builtin_amdgcn_readfirstlane(threadIdx.x >> 5);
  const int half = lane >> 4;
  const int l16  = lane & 15;
  const int n0   = (blockIdx.x * 4 + wave) * 16;

  __builtin_prefetch(fcW + (size_t)(n0 + l16) * HDIM, 0, 3);

  v8f acc[2] = {};
  for (int kc = 0; kc < 32; ++kc) {
    const int kbase = kc * 32;
    v16bf a0 = load_a_bf(h_bf + (0 * 16 + l16) * HDIM + kbase, half);
    v16bf a1 = load_a_bf(h_bf + (1 * 16 + l16) * HDIM + kbase, half);
    v16bf b  = load_b_bf(fcW + (size_t)(n0 + l16) * HDIM + kbase, half);
    acc[0] = __builtin_amdgcn_wmma_f32_16x16x32_bf16(false, a0, false, b, (short)0, acc[0], false, false);
    acc[1] = __builtin_amdgcn_wmma_f32_16x16x32_bf16(false, a1, false, b, (short)0, acc[1], false, false);
  }
  const int col = n0 + l16;
  const float bv = fc_b[col];
#pragma unroll
  for (int mt = 0; mt < 2; ++mt)
#pragma unroll
    for (int v = 0; v < 8; ++v) {
      const int row = mt * 16 + 8 * half + v;
      out[((size_t)row * TGT_LEN + t) * VOCAB + col] = acc[mt][v] + bv;
    }
}

// ---- argmax over V=32000 (first-index tie break) + teacher forcing select ----
__global__ __launch_bounds__(256) void argmax_select_kernel(
    const float* __restrict__ out, const int* __restrict__ X_tgt,
    const unsigned char* __restrict__ tf_mask, int t, int* __restrict__ cur_tok)
{
  const int b   = blockIdx.x;
  const int tid = threadIdx.x;
  const float* row = out + ((size_t)b * TGT_LEN + t) * VOCAB;
  float best = -INFINITY; int bidx = VOCAB;
  for (int i = tid; i < VOCAB; i += 256) {
    float v = row[i];
    if (v > best || (v == best && i < bidx)) { best = v; bidx = i; }
  }
  __shared__ float sv[256];
  __shared__ int   si[256];
  sv[tid] = best; si[tid] = bidx;
  __syncthreads();
  for (int s = 128; s > 0; s >>= 1) {
    if (tid < s) {
      float v2 = sv[tid + s]; int i2 = si[tid + s];
      if (v2 > sv[tid] || (v2 == sv[tid] && i2 < si[tid])) { sv[tid] = v2; si[tid] = i2; }
    }
    __syncthreads();
  }
  if (tid == 0) cur_tok[b] = tf_mask[t] ? X_tgt[b * TGT_LEN + t] : si[0];
}

// ---- one-time prep ----
__global__ void prep_combined_w(const float* __restrict__ Wih,   // [4096,512]
                                const float* __restrict__ Whh,   // [4096,1024]
                                __bf16* __restrict__ Wout)       // [4096,1536]
{
  const size_t n = (size_t)G4H * KDIM;
  for (size_t i = (size_t)blockIdx.x * blockDim.x + threadIdx.x; i < n;
       i += (size_t)gridDim.x * blockDim.x) {
    const size_t r = i / KDIM, c = i % KDIM;
    const float v = (c < EDIM) ? Wih[r * EDIM + c] : Whh[r * HDIM + (c - EDIM)];
    Wout[i] = (__bf16)v;
  }
}
__global__ void cvt_f32_bf16(const float* __restrict__ src, __bf16* __restrict__ dst, size_t n) {
  for (size_t i = (size_t)blockIdx.x * blockDim.x + threadIdx.x; i < n;
       i += (size_t)gridDim.x * blockDim.x)
    dst[i] = (__bf16)src[i];
}
__global__ void add_bias_kernel(const float* __restrict__ a, const float* __restrict__ b,
                                float* __restrict__ o, int n) {
  int i = blockIdx.x * blockDim.x + threadIdx.x;
  if (i < n) o[i] = a[i] + b[i];
}
__global__ void init_state_kernel(__bf16* hA, float* c, int* cur_tok) {
  int i = blockIdx.x * blockDim.x + threadIdx.x;
  if (i < BATCH * HDIM) { hA[i] = (__bf16)0.0f; c[i] = 0.0f; }
  if (i < BATCH) cur_tok[i] = SOS;
}

extern "C" void kernel_launch(void* const* d_in, const int* in_sizes, int n_in,
                              void* d_out, int out_size, void* d_ws, size_t ws_size,
                              hipStream_t stream) {
  (void)in_sizes; (void)n_in; (void)out_size; (void)ws_size;
  const int*   X_src   = (const int*)d_in[0];
  const int*   X_tgt   = (const int*)d_in[1];
  const unsigned char* tf_mask = (const unsigned char*)d_in[2];
  const float* src_emb = (const float*)d_in[3];
  const float* enc_Wih = (const float*)d_in[4];
  const float* enc_Whh = (const float*)d_in[5];
  const float* enc_bih = (const float*)d_in[6];
  const float* enc_bhh = (const float*)d_in[7];
  const float* tgt_emb = (const float*)d_in[8];
  const float* dec_Wih = (const float*)d_in[9];
  const float* dec_Whh = (const float*)d_in[10];
  const float* dec_bih = (const float*)d_in[11];
  const float* dec_bhh = (const float*)d_in[12];
  const float* fc_W    = (const float*)d_in[13];
  const float* fc_b    = (const float*)d_in[14];
  float* out = (float*)d_out;

  char* ws = (char*)d_ws;
  size_t off = 0;
  auto alloc = [&](size_t bytes) -> char* {
    char* p = ws + off;
    off = (off + bytes + 255) & ~(size_t)255;
    return p;
  };
  __bf16* W_enc  = (__bf16*)alloc((size_t)G4H * KDIM * 2);    // 12.6 MB
  __bf16* W_dec  = (__bf16*)alloc((size_t)G4H * KDIM * 2);    // 12.6 MB
  __bf16* fcWb   = (__bf16*)alloc((size_t)VOCAB * HDIM * 2);  // 65.5 MB (L2-resident)
  float*  b_enc  = (float*)alloc((size_t)G4H * 4);
  float*  b_dec  = (float*)alloc((size_t)G4H * 4);
  __bf16* hA     = (__bf16*)alloc((size_t)BATCH * HDIM * 2);  // h ping-pong
  __bf16* hB     = (__bf16*)alloc((size_t)BATCH * HDIM * 2);
  float*  c_st   = (float*)alloc((size_t)BATCH * HDIM * 4);
  int*    cur_tok = (int*)alloc((size_t)BATCH * 4);

  prep_combined_w<<<1024, 256, 0, stream>>>(enc_Wih, enc_Whh, W_enc);
  prep_combined_w<<<1024, 256, 0, stream>>>(dec_Wih, dec_Whh, W_dec);
  cvt_f32_bf16<<<2048, 256, 0, stream>>>(fc_W, fcWb, (size_t)VOCAB * HDIM);
  add_bias_kernel<<<(G4H + 255) / 256, 256, 0, stream>>>(enc_bih, enc_bhh, b_enc, G4H);
  add_bias_kernel<<<(G4H + 255) / 256, 256, 0, stream>>>(dec_bih, dec_bhh, b_dec, G4H);
  init_state_kernel<<<(BATCH * HDIM + 255) / 256, 256, 0, stream>>>(hA, c_st, cur_tok);

  const __bf16* h_in = hA;
  __bf16*       h_out = hB;
  // encoder
  for (int t = 0; t < SRC_LEN; ++t) {
    lstm_step_kernel<<<64, 128, 0, stream>>>(src_emb, X_src + t, SRC_LEN, W_enc, b_enc,
                                             h_in, h_out, c_st);
    const __bf16* tmp = h_out; h_out = (__bf16*)h_in; h_in = tmp;
  }
  // decoder (argmax feedback via cur_tok)
  for (int t = 0; t < TGT_LEN; ++t) {
    lstm_step_kernel<<<64, 128, 0, stream>>>(tgt_emb, cur_tok, 1, W_dec, b_dec,
                                             h_in, h_out, c_st);
    fc_logits_kernel<<<VOCAB / 64, 128, 0, stream>>>(h_out, fcWb, fc_b, out, t);
    argmax_select_kernel<<<BATCH, 256, 0, stream>>>(out, X_tgt, tf_mask, t, cur_tok);
    const __bf16* tmp = h_out; h_out = (__bf16*)h_in; h_in = tmp;
  }
}
